// EdgeConv2d_45320494907703
// MI455X (gfx1250) — compile-verified, matrix operand order checked by reading, not run on hardware
//
#include <hip/hip_runtime.h>

typedef __attribute__((ext_vector_type(16))) _Float16 v16h;
typedef __attribute__((ext_vector_type(8)))  _Float16 v8h;
typedef __attribute__((ext_vector_type(8)))  float    v8f;

#define BB    4
#define CC    64
#define NN    8192
#define KK    16
#define COUT  64
#define WAVES 8
#define WROW  136   // padded LDS row stride in halves (128 + 8) -> 272B, 16B aligned
#define FROW  136

__global__ __launch_bounds__(WAVES * 32)
void edgeconv_wmma_kernel(const float* __restrict__ x,
                          const int*   __restrict__ eidx,
                          const float* __restrict__ W,
                          const float* __restrict__ bias,
                          float*       __restrict__ out)
{
    __shared__ __align__(16) _Float16 Wl[COUT * WROW];            // 17408 B
    __shared__ __align__(16) _Float16 fT[WAVES][KK * FROW];       // 34816 B

    const int tid = threadIdx.x;

    // ---- cooperatively stage W (64 x 128 f32) as f16 into padded LDS ----
    for (int i = tid; i < COUT * 2 * CC; i += WAVES * 32) {
        const int r = i >> 7;          // /128
        const int c = i & 127;
        Wl[r * WROW + c] = (_Float16)W[i];
    }
    __syncthreads();

    const int wave = tid >> 5;
    const int lane = tid & 31;
    const int node = blockIdx.x * WAVES + wave;   // grid sized exactly B*N/WAVES
    const int b_   = node >> 13;                  // / 8192
    const int n_   = node & (NN - 1);

    // ---- edge indices: lanes 0-15 load neighbor (e=0), lanes 16-31 central (e=1)
    const int kq   = lane & 15;
    const int esel = lane >> 4;
    const int myidx = eidx[esel * (BB * NN * KK) + b_ * (NN * KK) + n_ * KK + kq];
    const int ij = __shfl(myidx, kq, 32);        // neighbor index for k = kq
    const int ii = __shfl(myidx, 16 + kq, 32);   // central  index for k = kq

    // ---- gather + build transposed feature tile featT[k][c] in f16 ----
    // lane handles (k = kq, channel half = esel): channels c0..c0+31
    const float*    xb   = x + b_ * (CC * NN);
    _Float16*       frow = &fT[wave][kq * FROW];
    const int       c0   = esel * 32;
    #pragma unroll 4
    for (int c = 0; c < 32; c += 2) {
        const float xi0 = xb[(c0 + c    ) * NN + ii];
        const float xi1 = xb[(c0 + c + 1) * NN + ii];
        const float xj0 = xb[(c0 + c    ) * NN + ij];
        const float xj1 = xb[(c0 + c + 1) * NN + ij];
        union { _Float16 h[2]; unsigned u; } p0, p1;
        p0.h[0] = (_Float16)xi0;         p0.h[1] = (_Float16)xi1;
        p1.h[0] = (_Float16)(xj0 - xi0); p1.h[1] = (_Float16)(xj1 - xi1);
        *(unsigned*)&frow[c0 + c]      = p0.u;   // rows 0..63   : x_i
        *(unsigned*)&frow[64 + c0 + c] = p1.u;   // rows 64..127 : x_j - x_i
    }

    // wave-local LDS visibility: drain this wave's DS stores (CDNA5 split counter)
    asm volatile("s_wait_dscnt 0" ::: "memory");

    // ---- WMMA: D[64 x 16] = W[64 x 128] * feat[128 x 16], f16 in / f32 acc ----
    const int hl  = lane >> 4;   // half-wave selects K sub-lane layout
    const int l15 = lane & 15;
    v8f acc[4] = {v8f{}, v8f{}, v8f{}, v8f{}};
    const _Float16* brow = &fT[wave][l15 * FROW];

    #pragma unroll
    for (int kc = 0; kc < 4; ++kc) {
        const int kb = kc * 32 + hl * 8;
        // B fragment (32x16): lane l15 = column, halves per ISA 16-bit layout
        const v8h b0 = *(const v8h*)&brow[kb];
        const v8h b1 = *(const v8h*)&brow[kb + 16];
        const v16h bf = __builtin_shufflevector(b0, b1,
            0, 1, 2, 3, 4, 5, 6, 7, 8, 9, 10, 11, 12, 13, 14, 15);
        #pragma unroll
        for (int m = 0; m < 4; ++m) {
            // A fragment (16x32): row = m*16 + l15 of W
            const _Float16* arow = &Wl[(m * 16 + l15) * WROW + kb];
            const v8h a0 = *(const v8h*)&arow[0];
            const v8h a1 = *(const v8h*)&arow[16];
            const v16h af = __builtin_shufflevector(a0, a1,
                0, 1, 2, 3, 4, 5, 6, 7, 8, 9, 10, 11, 12, 13, 14, 15);
            acc[m] = __builtin_amdgcn_wmma_f32_16x16x32_f16(
                false, af, false, bf, (short)0, acc[m], false, false);
        }
    }

    // ---- epilogue: +bias, max over 16 neighbor columns, ReLU, store ----
    const int rbase = hl * 8;    // C/D layout: VGPR v -> row v + 8*(lane>=16)
    #pragma unroll
    for (int m = 0; m < 4; ++m) {
        #pragma unroll
        for (int v = 0; v < 8; ++v) {
            const int row = m * 16 + rbase + v;
            float val = acc[m][v] + bias[row];
            // butterfly max across the 16 lanes of this half-wave (= 16 neighbors)
            #pragma unroll
            for (int mask = 1; mask <= 8; mask <<= 1)
                val = fmaxf(val, __shfl_xor(val, mask, 32));
            val = fmaxf(val, 0.0f);              // ReLU (monotone: ok after max)
            if (l15 == 0)
                out[(b_ * COUT + row) * NN + n_] = val;
        }
    }
}

extern "C" void kernel_launch(void* const* d_in, const int* in_sizes, int n_in,
                              void* d_out, int out_size, void* d_ws, size_t ws_size,
                              hipStream_t stream) {
    (void)in_sizes; (void)n_in; (void)out_size; (void)d_ws; (void)ws_size;
    const float* x    = (const float*)d_in[0];
    const int*   eidx = (const int*)  d_in[1];
    const float* W    = (const float*)d_in[2];
    const float* bias = (const float*)d_in[3];
    float*       out  = (float*)d_out;

    dim3 grid((BB * NN) / WAVES);   // 4096 blocks, 1 node per wave
    dim3 block(WAVES * 32);         // 8 wave32 waves
    hipLaunchKernelGGL(edgeconv_wmma_kernel, grid, block, 0, stream,
                       x, eidx, W, bias, out);
}
